// Head_17016660427122
// MI455X (gfx1250) — matrix operand order repaired and checked by reading, not compiled, not run on hardware
//
#include <hip/hip_runtime.h>
#include <hip/hip_bf16.h>
#include <hip/hip_fp16.h>

typedef __attribute__((ext_vector_type(16))) _Float16 v16h;
typedef __attribute__((ext_vector_type(8)))  _Float16 v8h;
typedef __attribute__((ext_vector_type(8)))  float    v8f;

#define WMMA_F16(a, b, c) \
  __builtin_amdgcn_wmma_f32_16x16x32_f16(false, (a), false, (b), (short)0, (c), false, false)

static constexpr int BB = 4;
static constexpr int T  = 2048;
static constexpr int C  = 1024;
static constexpr int H  = 64;

// ---------------------------------------------------------------------------
// Kernel 0: Wk/Wq/Wv [C][H] f32  ->  Wt [3][H][C] f16 (transposed, contiguous-K)
// ---------------------------------------------------------------------------
__global__ void transpose_w(const float* __restrict__ Wk,
                            const float* __restrict__ Wq,
                            const float* __restrict__ Wv,
                            _Float16* __restrict__ Wt) {
  int idx = blockIdx.x * blockDim.x + threadIdx.x;
  if (idx >= 3 * H * C) return;
  int p = idx / (H * C);
  int rem = idx % (H * C);
  int h = rem / C;
  int c = rem % C;
  const float* W = (p == 0) ? Wk : ((p == 1) ? Wq : Wv);
  Wt[idx] = (_Float16)W[(size_t)c * H + h];
}

// ---------------------------------------------------------------------------
// Kernel 1: projections. One wave computes a 16x64 tile of K, Q, V.
//   Q,K stored row-major f16 [B*T][64]; V stored transposed f16 [B][64][T].
// ---------------------------------------------------------------------------
__global__ __launch_bounds__(32) void proj_kernel(
    const float* __restrict__ x,          // [B*T][C] f32
    const _Float16* __restrict__ Wt,      // [3][H][C] f16
    _Float16* __restrict__ Kh,            // [B*T][H]
    _Float16* __restrict__ Qh,            // [B*T][H]
    _Float16* __restrict__ Vt)            // [B][H][T]
{
  const int tile = blockIdx.x;            // 16-row tile of flattened B*T
  const int lane = threadIdx.x;
  const int m  = lane & 15;
  const int hh = lane >> 4;
  const int kbaseA = hh * 8;              // A-layout K chunk base
  const int khalfB = hh * 16;             // B-layout K half base
  const int row0 = tile * 16;

  const float* xrow = x + (size_t)(row0 + m) * C;

  v8f acc[3][4] = {};

  for (int ks = 0; ks < C; ks += 32) {
    // A operand: x rows, f32 -> f16, contiguous K (k = 16*hh + s), same as B
    v8f f0 = *(const v8f*)(xrow + ks + khalfB);
    v8f f1 = *(const v8f*)(xrow + ks + khalfB + 8);
    union { v16h v; v8h h[2]; } au;
    au.h[0] = __builtin_convertvector(f0, v8h);
    au.h[1] = __builtin_convertvector(f1, v8h);
    const v16h a = au.v;

#pragma unroll
    for (int p = 0; p < 3; ++p) {
      const _Float16* wp = Wt + (size_t)p * H * C;
#pragma unroll
      for (int nt = 0; nt < 4; ++nt) {
        const int n = nt * 16 + m;
        const v16h b = *(const v16h*)(wp + (size_t)n * C + ks + khalfB);
        acc[p][nt] = WMMA_F16(a, b, acc[p][nt]);
      }
    }
  }

  // C/D layout: VGPR r, lane -> row M = r + 8*hh, col N = m
#pragma unroll
  for (int nt = 0; nt < 4; ++nt) {
#pragma unroll
    for (int r = 0; r < 8; ++r) {
      const int row = row0 + r + 8 * hh;       // flattened b*T + t
      const int col = nt * 16 + m;
      Kh[(size_t)row * H + col] = (_Float16)acc[0][nt][r];
      Qh[(size_t)row * H + col] = (_Float16)acc[1][nt][r];
      const int bidx = row >> 11;              // T = 2048
      const int tidx = row & 2047;
      Vt[((size_t)bidx * H + col) * T + tidx] = (_Float16)acc[2][nt][r];
    }
  }
}

// ---------------------------------------------------------------------------
// Kernel 2: causal flash attention, one wave per 16-query tile.
// ---------------------------------------------------------------------------
__global__ __launch_bounds__(32) void attn_kernel(
    const _Float16* __restrict__ Qh,      // [B*T][H]
    const _Float16* __restrict__ Kh,      // [B*T][H]
    const _Float16* __restrict__ Vt,      // [B][H][T]
    float* __restrict__ out)              // [B][T][H] f32
{
  __shared__ _Float16 Plds[16 * 32];      // P tile, row-major [16][32]

  const int gid = blockIdx.x;             // B * (T/16)
  const int b   = gid >> 7;               // T/16 = 128
  const int qt  = gid & 127;
  const int lane = threadIdx.x;
  const int m  = lane & 15;
  const int hh = lane >> 4;
  const int kbaseA = hh * 8;
  const int khalfB = hh * 16;

  // Load Q tile in A layout: H=64 -> two K=32 slices
  const _Float16* qrow = Qh + ((size_t)b * T + qt * 16 + m) * H;
  v16h qa[2];
#pragma unroll
  for (int ks = 0; ks < 2; ++ks) {
    union { v16h v; v8h h[2]; } qu;
    qu.h[0] = *(const v8h*)(qrow + ks * 32 + khalfB);
    qu.h[1] = *(const v8h*)(qrow + ks * 32 + khalfB + 8);
    qa[ks] = qu.v;
  }

  float mrow[8], ssum[8];
#pragma unroll
  for (int r = 0; r < 8; ++r) { mrow[r] = -__builtin_inff(); ssum[r] = 0.0f; }
  v8f O[4] = {};

  const float scale = 0.03125f;           // C**-0.5 = 1024^-0.5 (reference quirk)
  const int smax = qt * 16 + 15;

  for (int s0 = 0; s0 <= smax; s0 += 32) {
    // ---- S = Q K^T for two 16-key tiles (K reads are contiguous B-layout) ----
    v8f S0 = {}, S1 = {};
    const _Float16* k0 = Kh + ((size_t)b * T + s0 + m) * H;
    const _Float16* k1 = Kh + ((size_t)b * T + s0 + 16 + m) * H;
#pragma unroll
    for (int ks = 0; ks < 2; ++ks) {
      const v16h bk0 = *(const v16h*)(k0 + ks * 32 + khalfB);
      const v16h bk1 = *(const v16h*)(k1 + ks * 32 + khalfB);
      S0 = WMMA_F16(qa[ks], bk0, S0);
      S1 = WMMA_F16(qa[ks], bk1, S1);
    }

    // ---- scale + causal mask + online softmax (rows live per half-wave) ----
    float alpha[8];
#pragma unroll
    for (int r = 0; r < 8; ++r) {
      const int qg = qt * 16 + r + 8 * hh;
      float s0v = S0[r] * scale;
      float s1v = S1[r] * scale;
      if (s0 + m      > qg) s0v = -__builtin_inff();
      if (s0 + 16 + m > qg) s1v = -__builtin_inff();

      float mx = fmaxf(s0v, s1v);
#pragma unroll
      for (int off = 1; off < 16; off <<= 1)
        mx = fmaxf(mx, __shfl_xor(mx, off, 32));
      const float mnew = fmaxf(mrow[r], mx);
      const float a = __expf(mrow[r] - mnew);
      const float p0 = __expf(s0v - mnew);
      const float p1 = __expf(s1v - mnew);
      float rs = p0 + p1;
#pragma unroll
      for (int off = 1; off < 16; off <<= 1)
        rs += __shfl_xor(rs, off, 32);
      ssum[r] = ssum[r] * a + rs;
      mrow[r] = mnew;
      alpha[r] = a;
      S0[r] = p0;
      S1[r] = p1;
    }

    // rescale accumulators by alpha (per-row)
#pragma unroll
    for (int nt = 0; nt < 4; ++nt)
#pragma unroll
      for (int r = 0; r < 8; ++r)
        O[nt][r] *= alpha[r];

    // ---- bounce P through LDS: C-layout -> row-major [16][32] f16 ----
    __syncthreads();
#pragma unroll
    for (int r = 0; r < 8; ++r) {
      Plds[(r + 8 * hh) * 32 + m]      = (_Float16)S0[r];
      Plds[(r + 8 * hh) * 32 + 16 + m] = (_Float16)S1[r];
    }
    __syncthreads();

    // read back in A layout (16x32, K = key offset within step)
    union { v16h v; v8h h[2]; } pu;
    pu.h[0] = *(const v8h*)(&Plds[m * 32 + khalfB]);
    pu.h[1] = *(const v8h*)(&Plds[m * 32 + khalfB + 8]);
    const v16h pa = pu.v;

    // ---- O += P * V (V^T rows give contiguous B-layout loads) ----
#pragma unroll
    for (int nt = 0; nt < 4; ++nt) {
      const int n = nt * 16 + m;
      const v16h bv = *(const v16h*)(Vt + ((size_t)b * H + n) * T + s0 + khalfB);
      O[nt] = WMMA_F16(pa, bv, O[nt]);
    }
  }

  // ---- finalize: divide by row sums, store fp32 ----
#pragma unroll
  for (int r = 0; r < 8; ++r) {
    const float inv = 1.0f / ssum[r];
    const int qg = qt * 16 + r + 8 * hh;
#pragma unroll
    for (int nt = 0; nt < 4; ++nt)
      out[((size_t)b * T + qg) * H + nt * 16 + m] = O[nt][r] * inv;
  }
}

// ---------------------------------------------------------------------------
extern "C" void kernel_launch(void* const* d_in, const int* in_sizes, int n_in,
                              void* d_out, int out_size, void* d_ws, size_t ws_size,
                              hipStream_t stream) {
  (void)in_sizes; (void)n_in; (void)out_size; (void)ws_size;
  const float* x  = (const float*)d_in[0];
  const float* Wk = (const float*)d_in[1];
  const float* Wq = (const float*)d_in[2];
  const float* Wv = (const float*)d_in[3];
  float* out = (float*)d_out;

  // workspace layout (f16 elements)
  _Float16* Wt = (_Float16*)d_ws;                 // [3][H][C]
  _Float16* Qh = Wt + (size_t)3 * H * C;          // [B*T][H]
  _Float16* Kh = Qh + (size_t)BB * T * H;         // [B*T][H]
  _Float16* Vt = Kh + (size_t)BB * T * H;         // [B][H][T]

  const int nW = 3 * H * C;
  transpose_w<<<(nW + 255) / 256, 256, 0, stream>>>(Wk, Wq, Wv, Wt);

  const int nTiles = BB * T / 16;                 // 512
  proj_kernel<<<nTiles, 32, 0, stream>>>(x, Wt, Kh, Qh, Vt);
  attn_kernel<<<nTiles, 32, 0, stream>>>(Qh, Kh, Vt, out);
}
